// Attention_73083163509240
// MI455X (gfx1250) — matrix operand order repaired and checked by reading, not compiled, not run on hardware
//
#include <hip/hip_runtime.h>

typedef _Float16 half8 __attribute__((ext_vector_type(8)));
typedef _Float16 v16h  __attribute__((ext_vector_type(16)));
typedef float    v8f   __attribute__((ext_vector_type(8)));

#define WMMA_F32_F16(a,b,c) \
  __builtin_amdgcn_wmma_f32_16x16x32_f16(false,(a),false,(b),(short)0,(c),false,false)

constexpr int Bb = 2, T = 2048, D = 1024, H = 16, HS = 64;
constexpr int M  = Bb * T;               // 4096 rows for all projections

// Build the 16x32 f16 A-operand for one lane.
// Lane L (row = L&15) holds K = kb+0..15 in v[0..15], where kb = (L>>4)*16
// (contiguous order, matching the B loads). Two contiguous half8 loads.
__device__ __forceinline__ v16h load_a16(const _Float16* rowk0, int lh) {
  half8 lo = *(const half8*)(rowk0 + lh * 16);
  half8 hi = *(const half8*)(rowk0 + lh * 16 + 8);
  return __builtin_shufflevector(lo, hi, 0,1,2,3,4,5,6,7,8,9,10,11,12,13,14,15);
}

// ---------------- conversion kernels ----------------
__global__ void conv_f32_to_f16(const float* __restrict__ in,
                                _Float16* __restrict__ out, int n) {
  int i = blockIdx.x * blockDim.x + threadIdx.x;
  if (i < n) out[i] = (_Float16)in[i];
}

// W[K=1024][N=1024] f32 -> Wt[N][K] f16 (so B-operand loads are contiguous)
__global__ void conv_transpose_f16(const float* __restrict__ w,
                                   _Float16* __restrict__ wt) {
  int i = blockIdx.x * blockDim.x + threadIdx.x;   // over 1024*1024
  int kk = i >> 10, n = i & 1023;
  wt[(size_t)n * D + kk] = (_Float16)w[i];
}

// ---------------- WMMA GEMM: out = A[M,K] * Bt[N,K]^T + bias ----------------
// MODE 0: f16 out row-major [M,N]           (Q, K projections)
// MODE 1: f16 out scattered to V^T [B,H,HS,T] (V projection)
// MODE 2: f32 out row-major [M,N]           (output projection)
template <int MODE>
__global__ __launch_bounds__(256) void gemm_wmma(const _Float16* __restrict__ A,
                                                 const _Float16* __restrict__ Bt,
                                                 const float* __restrict__ bias,
                                                 void* __restrict__ out) {
  const int wv = threadIdx.x >> 5, lane = threadIdx.x & 31;
  const int ln = lane & 15, lh = lane >> 4;
  const int tid = blockIdx.x * 8 + wv;          // 4096 wave-tiles total
  const int m0 = (tid >> 4) << 4;               // 256 M-tiles
  const int n0 = (tid & 15) << 6;               // 16 N-tiles of 64

  v8f c0 = {}, c1 = {}, c2 = {}, c3 = {};
  const _Float16* arow = A + (size_t)(m0 + ln) * D;
  const _Float16* b0p = Bt + (size_t)(n0 +  0 + ln) * D + lh * 16;
  const _Float16* b1p = Bt + (size_t)(n0 + 16 + ln) * D + lh * 16;
  const _Float16* b2p = Bt + (size_t)(n0 + 32 + ln) * D + lh * 16;
  const _Float16* b3p = Bt + (size_t)(n0 + 48 + ln) * D + lh * 16;

  for (int k0 = 0; k0 < D; k0 += 32) {
    v16h a  = load_a16(arow + k0, lh);
    v16h b0 = *(const v16h*)(b0p + k0);
    v16h b1 = *(const v16h*)(b1p + k0);
    v16h b2 = *(const v16h*)(b2p + k0);
    v16h b3 = *(const v16h*)(b3p + k0);
    c0 = WMMA_F32_F16(a, b0, c0);
    c1 = WMMA_F32_F16(a, b1, c1);
    c2 = WMMA_F32_F16(a, b2, c2);
    c3 = WMMA_F32_F16(a, b3, c3);
  }

  v8f cc[4] = {c0, c1, c2, c3};
  #pragma unroll
  for (int t = 0; t < 4; ++t) {
    const int n = n0 + t * 16 + ln;
    const float bn = bias[n];
    #pragma unroll
    for (int i = 0; i < 8; ++i) {
      const int m = m0 + i + 8 * lh;           // C-layout: vgpr i, half-wave lh
      const float val = cc[t][i] + bn;
      if (MODE == 0) {
        ((_Float16*)out)[(size_t)m * D + n] = (_Float16)val;
      } else if (MODE == 1) {
        const int b = m >> 11, tp = m & (T - 1);
        const int h = n >> 6, d = n & 63;
        ((_Float16*)out)[((size_t)(b * H + h) * HS + d) * T + tp] = (_Float16)val;
      } else {
        ((float*)out)[(size_t)m * D + n] = val;
      }
    }
  }
}

// ---------------- RoPE (in-place on f16 q,k) ----------------
__global__ void rope_kernel(_Float16* __restrict__ q, _Float16* __restrict__ k) {
  size_t i = (size_t)blockIdx.x * blockDim.x + threadIdx.x;  // Bb*T*H*32 = 2^21
  if (i >= (size_t)Bb * T * H * 32) return;
  const int d = i & 31;
  const int h = (i >> 5) & 15;
  const int t = (i >> 9) & (T - 1);
  const int b = (int)(i >> 20);
  const size_t base = ((size_t)(b * T + t)) * D + h * 64 + d;
  const float invf = powf(10000.0f, -(float)d * (1.0f / 32.0f));
  const float ang = (float)t * invf;
  const float c = cosf(ang), s = sinf(ang);
  float q1 = (float)q[base], q2 = (float)q[base + 32];
  q[base]      = (_Float16)(q1 * c - q2 * s);
  q[base + 32] = (_Float16)(q2 * c + q1 * s);
  float k1 = (float)k[base], k2 = (float)k[base + 32];
  k[base]      = (_Float16)(k1 * c - k2 * s);
  k[base + 32] = (_Float16)(k2 * c + k1 * s);
}

// ---------------- flash attention: one wave per 16-query tile ----------------
__global__ __launch_bounds__(256) void flash_attn(const _Float16* __restrict__ q,
                                                  const _Float16* __restrict__ k,
                                                  const _Float16* __restrict__ vt,
                                                  const int* __restrict__ amask,
                                                  _Float16* __restrict__ y) {
  __shared__ _Float16 plds[8 * 16 * 32];        // 1 KB P-tile per wave
  const int wv = threadIdx.x >> 5, lane = threadIdx.x & 31;
  const int ln = lane & 15, lh = lane >> 4;
  const int wid = blockIdx.x * 8 + wv;          // 4096 waves total
  const int qt = wid & 127, bh = wid >> 7;
  const int h = bh & 15, b = bh >> 4;
  const int q0 = qt * 16;
  _Float16* P = plds + wv * 512;

  const size_t headoff = (size_t)h * HS;
  const _Float16* qrow = q + ((size_t)(b * T) + q0 + ln) * D + headoff;
  v16h qa0 = load_a16(qrow, lh);                // d = 0..31
  v16h qa1 = load_a16(qrow + 32, lh);           // d = 32..63

  float mrow[8], lrow[8];
  #pragma unroll
  for (int i = 0; i < 8; ++i) { mrow[i] = -3.0e38f; lrow[i] = 0.0f; }
  v8f o0 = {}, o1 = {}, o2 = {}, o3 = {};

  const _Float16* kbase = k + (size_t)(b * T) * D + headoff;
  const _Float16* vbase = vt + (size_t)(b * H + h) * HS * T;
  const int* mk = amask + b * T;
  const float NEG = -3.0e38f;

  for (int jb = 0; jb < q0 + 16; jb += 32) {    // 32 keys per chunk
    const int ki0 = jb + ln, ki1 = jb + 16 + ln;
    const _Float16* k0p = kbase + (size_t)ki0 * D;
    const _Float16* k1p = kbase + (size_t)ki1 * D;
    v16h kb00 = *(const v16h*)(k0p + lh * 16);        // d 0..31, col = key ki0
    v16h kb01 = *(const v16h*)(k0p + 32 + lh * 16);   // d 32..63
    v16h kb10 = *(const v16h*)(k1p + lh * 16);
    v16h kb11 = *(const v16h*)(k1p + 32 + lh * 16);

    v8f s0 = {}, s1 = {};
    s0 = WMMA_F32_F16(qa0, kb00, s0);
    s0 = WMMA_F32_F16(qa1, kb01, s0);
    s1 = WMMA_F32_F16(qa0, kb10, s1);
    s1 = WMMA_F32_F16(qa1, kb11, s1);

    const bool am0 = mk[ki0] != 0, am1 = mk[ki1] != 0;
    #pragma unroll
    for (int i = 0; i < 8; ++i) {
      const int qi = q0 + i + 8 * lh;           // C-layout row for this vgpr
      float r0 = s0[i] * 0.125f;                // 1/sqrt(64)
      float r1 = s1[i] * 0.125f;
      if (!am0 || ki0 > qi) r0 = NEG;           // causal + attention_mask
      if (!am1 || ki1 > qi) r1 = NEG;
      // row max over the 16 lanes holding this row (xor<=8 stays in half-wave)
      float mx = fmaxf(r0, r1);
      mx = fmaxf(mx, __shfl_xor(mx, 1, 32));
      mx = fmaxf(mx, __shfl_xor(mx, 2, 32));
      mx = fmaxf(mx, __shfl_xor(mx, 4, 32));
      mx = fmaxf(mx, __shfl_xor(mx, 8, 32));
      const float mnew = fmaxf(mrow[i], mx);
      const float corr = __expf(mrow[i] - mnew);
      const float p0 = __expf(r0 - mnew);
      const float p1 = __expf(r1 - mnew);
      float rs = p0 + p1;
      rs += __shfl_xor(rs, 1, 32);
      rs += __shfl_xor(rs, 2, 32);
      rs += __shfl_xor(rs, 4, 32);
      rs += __shfl_xor(rs, 8, 32);
      lrow[i] = lrow[i] * corr + rs;
      mrow[i] = mnew;
      o0[i] *= corr; o1[i] *= corr; o2[i] *= corr; o3[i] *= corr;
      // C-layout -> LDS (row-major 16x32) for re-swizzle into A-layout
      P[(i + 8 * lh) * 32 + ln] = (_Float16)p0;
      P[(i + 8 * lh) * 32 + 16 + ln] = (_Float16)p1;
    }
    // wave-synchronous LDS reuse: DS ops are in-order per wave; fence compiler
    asm volatile("s_wait_dscnt 0x0" ::: "memory");
    v16h pa = load_a16(P + (size_t)ln * 32, lh);      // P as 16x32 A-operand

    const _Float16* vcol = vbase + (size_t)ln * T + jb + lh * 16;
    v16h vb0 = *(const v16h*)(vcol + 0 * 16 * T);     // B: col = out dim, K = key
    v16h vb1 = *(const v16h*)(vcol + 1 * 16 * T);
    v16h vb2 = *(const v16h*)(vcol + 2 * 16 * T);
    v16h vb3 = *(const v16h*)(vcol + 3 * 16 * T);
    o0 = WMMA_F32_F16(pa, vb0, o0);
    o1 = WMMA_F32_F16(pa, vb1, o1);
    o2 = WMMA_F32_F16(pa, vb2, o2);
    o3 = WMMA_F32_F16(pa, vb3, o3);
  }

  #pragma unroll
  for (int i = 0; i < 8; ++i) {
    const float inv = 1.0f / lrow[i];
    const int qi = q0 + i + 8 * lh;
    _Float16* yr = y + ((size_t)(b * T) + qi) * D + headoff;
    yr[ 0 + ln] = (_Float16)(o0[i] * inv);
    yr[16 + ln] = (_Float16)(o1[i] * inv);
    yr[32 + ln] = (_Float16)(o2[i] * inv);
    yr[48 + ln] = (_Float16)(o3[i] * inv);
  }
}

// ---------------- host ----------------
extern "C" void kernel_launch(void* const* d_in, const int* in_sizes, int n_in,
                              void* d_out, int out_size, void* d_ws, size_t ws_size,
                              hipStream_t stream) {
  const float* x  = (const float*)d_in[0];
  const int*   am = (const int*)d_in[1];
  const float* Wq = (const float*)d_in[2]; const float* bq = (const float*)d_in[3];
  const float* Wk = (const float*)d_in[4]; const float* bk = (const float*)d_in[5];
  const float* Wv = (const float*)d_in[6]; const float* bv = (const float*)d_in[7];
  const float* Wo = (const float*)d_in[8]; const float* bo = (const float*)d_in[9];

  char* ws = (char*)d_ws;
  const size_t actBytes = (size_t)M * D * sizeof(_Float16);   // 8 MiB
  const size_t wBytes   = (size_t)D * D * sizeof(_Float16);   // 2 MiB
  _Float16* xh  = (_Float16*)ws;            ws += actBytes;
  _Float16* wqT = (_Float16*)ws;            ws += wBytes;
  _Float16* wkT = (_Float16*)ws;            ws += wBytes;
  _Float16* wvT = (_Float16*)ws;            ws += wBytes;
  _Float16* woT = (_Float16*)ws;            ws += wBytes;
  _Float16* qh  = (_Float16*)ws;            ws += actBytes;
  _Float16* kh  = (_Float16*)ws;            ws += actBytes;
  _Float16* vt  = (_Float16*)ws;            ws += actBytes;   // [B,H,HS,T]
  _Float16* yh  = xh;                       // xh dead after QKV GEMMs; reuse

  conv_f32_to_f16<<<(M * D + 255) / 256, 256, 0, stream>>>(x, xh, M * D);
  conv_transpose_f16<<<(D * D) / 256, 256, 0, stream>>>(Wq, wqT);
  conv_transpose_f16<<<(D * D) / 256, 256, 0, stream>>>(Wk, wkT);
  conv_transpose_f16<<<(D * D) / 256, 256, 0, stream>>>(Wv, wvT);
  conv_transpose_f16<<<(D * D) / 256, 256, 0, stream>>>(Wo, woT);

  gemm_wmma<0><<<512, 256, 0, stream>>>(xh, wqT, bq, qh);
  gemm_wmma<0><<<512, 256, 0, stream>>>(xh, wkT, bk, kh);
  gemm_wmma<1><<<512, 256, 0, stream>>>(xh, wvT, bv, vt);

  rope_kernel<<<(Bb * T * H * 32 + 255) / 256, 256, 0, stream>>>(qh, kh);

  flash_attn<<<512, 256, 0, stream>>>(qh, kh, vt, am, yh);

  gemm_wmma<2><<<512, 256, 0, stream>>>(yh, woT, bo, d_out);
}
